// NonparametricCrossAttentionPooling_41188736369110
// MI455X (gfx1250) — matrix operand order repaired and checked by reading, not compiled, not run on hardware
//
#include <hip/hip_runtime.h>
#include <hip/hip_bf16.h>
#include <math.h>

// ---------------------------------------------------------------------------
// NonparametricCrossAttentionPooling for MI455X (gfx1250, wave32, WMMA)
// Pass 0: per-key squared norms (f32, one HBM sweep)
// Pass 1: fused  S=Q.K^T -> gaussian-mixture weights -> nf = W.K  (flash style,
//         double-buffered LDS K-tiles, register prefetch, 1 barrier / tile)
// Pass 2-3: deterministic per-feature mean/var reduction
// Pass 4: BatchNorm + exact GELU (in place on d_out)
// ---------------------------------------------------------------------------

typedef __attribute__((ext_vector_type(16))) _Float16 v16h;
typedef __attribute__((ext_vector_type(8)))  float    v8f;

union AFrag { v16h v; float4 q[2]; };
struct __align__(16) H8 { _Float16 h[8]; };

__device__ __forceinline__ v8f wmma_f16(v16h a, v16h b, v8f c) {
  // D = A(16x32 f16) * B(32x16 f16) + C(16x16 f32)
  return __builtin_amdgcn_wmma_f32_16x16x32_f16(false, a, false, b, (short)0, c,
                                                false, false);
}

#define BATCH 8
#define NQ 4096
#define NK 4096
#define FDIM 64
#define KT 32            // keys per tile
#define NKT (NK / KT)    // 128 tiles
#define KKF_STRIDE 80    // halves per row of K[k][f] tile (64 + pad, conflict-free)
#define KFK_STRIDE 40    // halves per row of K^T[f][k] tile (32 + pad)
#define W_STRIDE 40      // halves per row of W[q][k] per-wave buffer

// ---- pass 0: per-key squared norms in full f32 precision --------------------
__global__ __launch_bounds__(256) void nca_k2(const float* __restrict__ K,
                                              float* __restrict__ k2g) {
  const int r = blockIdx.x * 256 + threadIdx.x;   // BATCH*NK rows
  const float* kp = K + (size_t)r * FDIM;
  float s = 0.f;
#pragma unroll
  for (int i = 0; i < 16; ++i) {
    float4 x = *(const float4*)(kp + 4 * i);
    s += x.x * x.x + x.y * x.y + x.z * x.z + x.w * x.w;
  }
  k2g[r] = s;
}

__global__ __launch_bounds__(256) void nca_pass1(const float* __restrict__ Q,
                                                 const float* __restrict__ K,
                                                 const float* __restrict__ k2g,
                                                 float* __restrict__ out) {
  __shared__ _Float16 sKkf[2][KT * KKF_STRIDE];    // K tile [k][f]  (GEMM1 A)
  __shared__ _Float16 sKfk[2][FDIM * KFK_STRIDE];  // K tile [f][k]  (GEMM2 A)
  __shared__ __align__(16) float sK2[2][KT];       // per-key squared norms
  __shared__ _Float16 sW[8][16 * W_STRIDE];        // per-wave W[q][k]

  const int tid  = threadIdx.x;
  const int wave = tid >> 5;
  const int lane = tid & 31;
  const int h    = lane >> 4;   // lane half (WMMA layout)
  const int ln   = lane & 15;

  const int b    = blockIdx.x >> 5;                        // 32 blocks per batch
  const int qrow = (blockIdx.x & 31) * 128 + wave * 16 + ln;

  // cooperative K-tile addressing: 8 f32 per thread per tile
  const int kl = tid >> 3;
  const int f0 = (tid & 7) * 8;
  const float* kbase  = K + ((size_t)b * NK + kl) * FDIM + f0;
  const float* k2base = k2g + (size_t)b * NK;

  // ------- load Q as the two B operands (resident all kernel) + ||q||^2 -----
  const float* qp = Q + ((size_t)b * NQ + qrow) * FDIM;
  AFrag bq0, bq1;
  float q2p = 0.f;
#pragma unroll
  for (int i = 0; i < 16; ++i) {
    float x = qp[8 * h + (i & 7) + 2 * (i & 8)];        // B op0: features 8h+(i&7), +16 for i>=8
    float y = qp[32 + 8 * h + (i & 7) + 2 * (i & 8)];   // B op1: same, offset 32
    q2p += x * x + y * y;
    bq0.v[i] = (_Float16)x;
    bq1.v[i] = (_Float16)y;
  }
  const float q2 = q2p + __shfl_xor(q2p, 16, 32);   // full ||q||^2 (f32)

  const v8f zero = {0.f, 0.f, 0.f, 0.f, 0.f, 0.f, 0.f, 0.f};
  v8f acc[4];   // nf^T accumulators, 4 feature blocks of 16
#pragma unroll
  for (int fb = 0; fb < 4; ++fb) acc[fb] = zero;
  float wsum = 0.f;

  // convert a register-held K tile into the two LDS layouts + norm row
  auto stash = [&](int buf, float4 x0, float4 x1, float rk2) {
    float v[8] = {x0.x, x0.y, x0.z, x0.w, x1.x, x1.y, x1.z, x1.w};
    H8 pk;
#pragma unroll
    for (int i = 0; i < 8; ++i) pk.h[i] = (_Float16)v[i];
    *(float4*)&sKkf[buf][kl * KKF_STRIDE + f0] = *(const float4*)&pk;
#pragma unroll
    for (int i = 0; i < 8; ++i) sKfk[buf][(f0 + i) * KFK_STRIDE + kl] = pk.h[i];
    if (tid < KT) sK2[buf][tid] = rk2;   // wave 0 only (uniform per wave)
  };

  // ---- software pipeline prologue: tile 0 regs -> LDS buffer 0 --------------
  float4 r0 = *(const float4*)kbase;
  float4 r1 = *(const float4*)(kbase + 4);
  float rk2 = (tid < KT) ? k2base[tid] : 0.f;
  stash(0, r0, r1, rk2);
  __syncthreads();

  for (int kt = 0; kt < NKT; ++kt) {
    const int cur = kt & 1;
    const int nxt = cur ^ 1;

    // ---- prefetch next tile into registers (latency hidden under compute) --
    if (kt + 1 < NKT) {
      const float* kp = kbase + (size_t)(kt + 1) * KT * FDIM;
      r0 = *(const float4*)kp;
      r1 = *(const float4*)(kp + 4);
      if (tid < KT) rk2 = k2base[(kt + 1) * KT + tid];
    }
    if (kt + 2 < NKT)   // pull tile t+2 toward L2 (global_prefetch_b8)
      __builtin_prefetch(kbase + (size_t)(kt + 2) * KT * FDIM, 0, 1);

    // ---- GEMM1: S^T(16k x 16q) = Ktile * Q^T, two 16-key subtiles ----------
    const _Float16* kkf = sKkf[cur];
    AFrag a0, a1;
    a0.q[0] = *(const float4*)&kkf[ln * KKF_STRIDE + 8 * h];
    a0.q[1] = *(const float4*)&kkf[ln * KKF_STRIDE + 16 + 8 * h];
    a1.q[0] = *(const float4*)&kkf[ln * KKF_STRIDE + 32 + 8 * h];
    a1.q[1] = *(const float4*)&kkf[ln * KKF_STRIDE + 48 + 8 * h];
    v8f s0 = wmma_f16(a0.v, bq0.v, zero);
    s0 = wmma_f16(a1.v, bq1.v, s0);

    a0.q[0] = *(const float4*)&kkf[(16 + ln) * KKF_STRIDE + 8 * h];
    a0.q[1] = *(const float4*)&kkf[(16 + ln) * KKF_STRIDE + 16 + 8 * h];
    a1.q[0] = *(const float4*)&kkf[(16 + ln) * KKF_STRIDE + 32 + 8 * h];
    a1.q[1] = *(const float4*)&kkf[(16 + ln) * KKF_STRIDE + 48 + 8 * h];
    v8f s1 = wmma_f16(a0.v, bq0.v, zero);
    s1 = wmma_f16(a1.v, bq1.v, s1);

    // ---- gaussian-mixture weights in f32 (bw = 1.0) ------------------------
    H8 p0, p1;
#pragma unroll
    for (int i = 0; i < 8; ++i) {
      float qk2a = q2 + sK2[cur][8 * h + i];          // per-(lane, vgpr) const
      float qk2b = q2 + sK2[cur][16 + 8 * h + i];
      float n2 = fmaxf(qk2a - 2.f * s0[i], 1e-12f);
      float w = 0.5f * __expf(-0.5f * n2) + 0.3f * __expf(-0.125f * n2)
              + 0.2f * __expf(-2.f * n2);
      wsum += w; p0.h[i] = (_Float16)w;

      float n2b = fmaxf(qk2b - 2.f * s1[i], 1e-12f);
      float wb = 0.5f * __expf(-0.5f * n2b) + 0.3f * __expf(-0.125f * n2b)
               + 0.2f * __expf(-2.f * n2b);
      wsum += wb; p1.h[i] = (_Float16)wb;
    }

    // ---- D-layout -> B-layout transpose through per-wave LDS ---------------
    _Float16* wrow = &sW[wave][ln * W_STRIDE];
    *(float4*)&wrow[8 * h]      = *(const float4*)&p0;   // k = 8h..8h+7
    *(float4*)&wrow[16 + 8 * h] = *(const float4*)&p1;   // k = 16+8h..
    asm volatile("s_wait_dscnt 0" ::: "memory");         // same-wave DS order

    AFrag bw;
    bw.q[0] = *(const float4*)&wrow[8 * h];
    bw.q[1] = *(const float4*)&wrow[16 + 8 * h];

    // ---- GEMM2: nf^T(16f x 16q) += K^T block * W^T -------------------------
    const _Float16* kfk = sKfk[cur];
#pragma unroll
    for (int fb = 0; fb < 4; ++fb) {
      AFrag ak;
      ak.q[0] = *(const float4*)&kfk[(16 * fb + ln) * KFK_STRIDE + 8 * h];
      ak.q[1] = *(const float4*)&kfk[(16 * fb + ln) * KFK_STRIDE + 16 + 8 * h];
      acc[fb] = wmma_f16(ak.v, bw.v, acc[fb]);
    }

    // ---- stage next tile into the other LDS buffer, single barrier ---------
    if (kt + 1 < NKT) stash(nxt, r0, r1, rk2);
    __syncthreads();
  }

  // ---- normalize by row weight sum and store nf ----------------------------
  const float wtot = wsum + __shfl_xor(wsum, 16, 32);
  const float inv = 1.f / (wtot + 1e-8f);
  float* op = out + ((size_t)b * NQ + qrow) * FDIM;
#pragma unroll
  for (int fb = 0; fb < 4; ++fb) {
    float4 lo, hi;
    lo.x = acc[fb][0] * inv; lo.y = acc[fb][1] * inv;
    lo.z = acc[fb][2] * inv; lo.w = acc[fb][3] * inv;
    hi.x = acc[fb][4] * inv; hi.y = acc[fb][5] * inv;
    hi.z = acc[fb][6] * inv; hi.w = acc[fb][7] * inv;
    *(float4*)&op[16 * fb + 8 * h]     = lo;   // f = 16fb + 8h + {0..3}
    *(float4*)&op[16 * fb + 8 * h + 4] = hi;   // f = 16fb + 8h + {4..7}
  }
}

// ---- pass 2: deterministic per-feature partial sums (64 blocks x 512 rows) --
__global__ __launch_bounds__(256) void nca_stats(const float* __restrict__ nf,
                                                 float* __restrict__ ws) {
  __shared__ float rs[256], rq[256];
  const int t = threadIdx.x;
  const int f = t & 63;
  const int r0 = blockIdx.x * 512 + (t >> 6);
  float s = 0.f, ss = 0.f;
  for (int i = 0; i < 128; ++i) {
    float x = nf[(size_t)(r0 + 4 * i) * FDIM + f];
    s += x; ss += x * x;
  }
  rs[t] = s; rq[t] = ss;
  __syncthreads();
  if (t < 64) {
    float a  = rs[t] + rs[t + 64] + rs[t + 128] + rs[t + 192];
    float a2 = rq[t] + rq[t + 64] + rq[t + 128] + rq[t + 192];
    ws[blockIdx.x * 64 + t]        = a;
    ws[4096 + blockIdx.x * 64 + t] = a2;
  }
}

// ---- pass 3: finalize mean / biased var per feature -------------------------
__global__ void nca_stats2(float* __restrict__ ws) {
  const int f = threadIdx.x;   // 64 threads
  float s = 0.f, ss = 0.f;
  for (int i = 0; i < 64; ++i) {
    s  += ws[i * 64 + f];
    ss += ws[4096 + i * 64 + f];
  }
  const float inv_n = 1.f / (float)(BATCH * NQ);
  float mean = s * inv_n;
  float var  = ss * inv_n - mean * mean;
  ws[8192 + f] = mean;
  ws[8256 + f] = var;
}

// ---- pass 4: BatchNorm (training stats) + exact GELU, in place --------------
__global__ __launch_bounds__(256) void nca_bngelu(float* __restrict__ out,
                                                  const float* __restrict__ ws,
                                                  const float* __restrict__ gamma,
                                                  const float* __restrict__ beta) {
  const size_t idx = (size_t)blockIdx.x * 256 + threadIdx.x;
  const int f = (int)(idx & 63);
  float x = out[idx];
  float mean = ws[8192 + f];
  float var  = ws[8256 + f];
  float xh = (x - mean) * rsqrtf(var + 1e-5f);
  float y = xh * gamma[f] + beta[f];
  out[idx] = 0.5f * y * (1.f + erff(y * 0.70710678118654752f));
}

extern "C" void kernel_launch(void* const* d_in, const int* in_sizes, int n_in,
                              void* d_out, int out_size, void* d_ws, size_t ws_size,
                              hipStream_t stream) {
  const float* Q     = (const float*)d_in[0];
  const float* K     = (const float*)d_in[1];
  const float* gamma = (const float*)d_in[2];
  const float* beta  = (const float*)d_in[3];
  float* out = (float*)d_out;
  float* ws  = (float*)d_ws;
  // ws layout (floats): [0,4096) sum partials, [4096,8192) sq partials,
  // [8192,8256) mean, [8256,8320) var, [8320, 8320+BATCH*NK) key norms
  float* k2g = ws + 8320;

  nca_k2<<<(BATCH * NK) / 256, 256, 0, stream>>>(K, k2g);          // 128 blocks
  nca_pass1<<<BATCH * (NQ / 128), 256, 0, stream>>>(Q, K, k2g, out);
  nca_stats<<<64, 256, 0, stream>>>(out, ws);
  nca_stats2<<<1, 64, 0, stream>>>(ws);
  nca_bngelu<<<(BATCH * NQ * FDIM) / 256, 256, 0, stream>>>(out, ws, gamma, beta);
}
